// WorldSpaceTrianglePipeline_71365176590465
// MI455X (gfx1250) — matrix operand order repaired and checked by reading, not compiled, not run on hardware
//
#include <hip/hip_runtime.h>
#include <hip/hip_bf16.h>

typedef __attribute__((ext_vector_type(16))) _Float16 v16h;
typedef __attribute__((ext_vector_type(8)))  _Float16 v8h;
typedef __attribute__((ext_vector_type(8)))  float    v8f;

#define N_VERTS 100000
#define N_PIX   262144
#define FEAT    32
#define GS      128
#define DHW     (128*128*128)
#define KPAD_IN 96          // 89 valid, padded to multiple of 32
#define WIDTH   128

union HU { v16h v; v8h h[2]; };

// ---------------------------------------------------------------------------
// Stage 1: trilinear sample grid at each vertex -> vfeat [V][32] f32
// ---------------------------------------------------------------------------
__global__ __launch_bounds__(256) void k_vertex_feat(
    const float* __restrict__ verts, const float* __restrict__ grid,
    float* __restrict__ vfeat)
{
  int v = blockIdx.x * blockDim.x + threadIdx.x;
  if (v >= N_VERTS) return;
  float p0 = verts[v*3+0], p1 = verts[v*3+1], p2 = verts[v*3+2];
  int i0[3], i1[3]; float fr[3];
  float pp[3] = {p0, p1, p2};
  #pragma unroll
  for (int d = 0; d < 3; ++d) {
    // norm = 2v-1 ; t = (norm+1)*0.5*(size-1) = v*127
    float t  = pp[d] * 127.0f;
    float f0 = fminf(fmaxf(floorf(t), 0.0f), 127.0f);
    float f1 = fminf(f0 + 1.0f, 127.0f);
    i0[d] = (int)f0; i1[d] = (int)f1; fr[d] = t - f0;
  }
  int   offs[8];
  float w[8];
  #pragma unroll
  for (int cx = 0; cx < 2; ++cx) {
    int   xi = cx ? i1[0] : i0[0];
    float wx = cx ? fr[0] : 1.0f - fr[0];
    #pragma unroll
    for (int cy = 0; cy < 2; ++cy) {
      int   yi = cy ? i1[1] : i0[1];
      float wy = cy ? fr[1] : 1.0f - fr[1];
      #pragma unroll
      for (int cz = 0; cz < 2; ++cz) {
        int   zi = cz ? i1[2] : i0[2];
        float wz = cz ? fr[2] : 1.0f - fr[2];
        offs[cx*4+cy*2+cz] = (xi*GS + yi)*GS + zi;
        w[cx*4+cy*2+cz]    = wx*wy*wz;
      }
    }
  }
  for (int c = 0; c < FEAT; ++c) {
    const float* g = grid + (size_t)c * DHW;
    float s = 0.0f;
    #pragma unroll
    for (int k = 0; k < 8; ++k) s += w[k] * g[offs[k]];
    vfeat[(size_t)v*FEAT + c] = s;
  }
}

// ---------------------------------------------------------------------------
// Stage 2: per-pixel barycentric combine + pos-embed -> mlpin [P][96] f16
// ---------------------------------------------------------------------------
__global__ __launch_bounds__(256) void k_pixel_input(
    const float* __restrict__ verts, const int* __restrict__ pvi,
    const float* __restrict__ bary, const float* __restrict__ vfeat,
    _Float16* __restrict__ mlpin)
{
  int p = blockIdx.x * blockDim.x + threadIdx.x;
  if (p >= N_PIX) return;
  int ia = pvi[p*3+0], ib = pvi[p*3+1], ic = pvi[p*3+2];
  float ba = bary[p*3+0], bb = bary[p*3+1], bc = bary[p*3+2];
  _Float16* dst = mlpin + (size_t)p * KPAD_IN;
  const float* fa = vfeat + (size_t)ia * FEAT;
  const float* fb = vfeat + (size_t)ib * FEAT;
  const float* fc = vfeat + (size_t)ic * FEAT;
  for (int c = 0; c < FEAT; ++c)
    dst[c] = (_Float16)(ba*fa[c] + bb*fb[c] + bc*fc[c]);
  float co[3];
  #pragma unroll
  for (int d = 0; d < 3; ++d) {
    co[d] = ba*verts[ia*3+d] + bb*verts[ib*3+d] + bc*verts[ic*3+d];
    dst[32+d] = (_Float16)co[d];
  }
  #pragma unroll
  for (int k = 0; k < 9; ++k) {
    float sc = (float)(1 << k);
    #pragma unroll
    for (int d = 0; d < 3; ++d) {
      float x = co[d] * sc;
      dst[35 + k*6 + d] = (_Float16)sinf(x);
      dst[38 + k*6 + d] = (_Float16)cosf(x);
    }
  }
  #pragma unroll
  for (int i = 89; i < KPAD_IN; ++i) dst[i] = (_Float16)0.0f;
}

// ---------------------------------------------------------------------------
// Stage 3a: repack weights into WMMA B-fragment streams (f16)
// Layout per layer: [ntile][kstep][lane(32)][16 halves]
// B layout (32x16, f16): lanes 0-15: N=lane, halves=K 0..15 of the kstep;
//                        lanes 16-31: N=lane-16, halves=K 16..31.
// Skip layers (1,2,4): ksteps 0..2 read mlp_input rows (89 valid of 96),
//                      ksteps 3..6 read hidden rows (128).
// ---------------------------------------------------------------------------
__global__ __launch_bounds__(256) void k_prepack(
    const float* w0, const float* w1, const float* w2, const float* w3,
    const float* w4, const float* w5, const float* w6, const float* w7,
    const float* w8, _Float16* __restrict__ frags)
{
  const int kst[9] = {3,7,7,4,7,4,4,4,4};
  const int ntl[9] = {8,8,8,8,8,8,8,8,1};
  int layer = blockIdx.y;
  int tid   = blockIdx.x * 256 + threadIdx.x;
  int nfr   = ntl[layer] * kst[layer];
  if (tid >= nfr * 32) return;
  int lane = tid & 31, f = tid >> 5;
  int s = f % kst[layer], ntile = f / kst[layer];
  size_t off = 0;
  for (int i = 0; i < layer; ++i) off += (size_t)ntl[i] * kst[i] * 512;
  _Float16* dst = frags + off + (size_t)f * 512 + lane * 16;
  const float* W = w0; int din = 89; int outd = 128; int skip = 0;
  switch (layer) {
    case 0: W = w0; din = 89;  break;
    case 1: W = w1; din = 217; skip = 1; break;
    case 2: W = w2; din = 217; skip = 1; break;
    case 3: W = w3; din = 128; break;
    case 4: W = w4; din = 217; skip = 1; break;
    case 5: W = w5; din = 128; break;
    case 6: W = w6; din = 128; break;
    case 7: W = w7; din = 128; break;
    default: W = w8; din = 128; outd = 3; break;
  }
  int sel = lane >> 4, n = lane & 15, N = ntile*16 + n;
  #pragma unroll
  for (int h = 0; h < 16; ++h) {
    int kk = s*32 + sel*16 + h;
    int wrow; int valid = 1;
    if (skip) {
      if (kk < 96) { wrow = kk; valid = (kk < 89); }
      else         { wrow = 89 + (kk - 96); }
    } else { wrow = kk; valid = (kk < din); }
    float val = 0.0f;
    if (valid && N < outd) val = W[(size_t)wrow * outd + N];
    dst[h] = (_Float16)val;
  }
}

__global__ void k_biases(const float* b0, const float* b1, const float* b2,
                         const float* b3, const float* b4, const float* b5,
                         const float* b6, const float* b7, const float* rb,
                         float* __restrict__ dst)
{
  int tid = blockIdx.x * blockDim.x + threadIdx.x;
  if (tid < 1024) {
    const float* b = b0;
    switch (tid >> 7) {
      case 0: b = b0; break; case 1: b = b1; break; case 2: b = b2; break;
      case 3: b = b3; break; case 4: b = b4; break; case 5: b = b5; break;
      case 6: b = b6; break; default: b = b7; break;
    }
    dst[tid] = b[tid & 127];
  } else if (tid < 1027) {
    dst[tid] = rb[tid - 1024];
  }
}

// ---------------------------------------------------------------------------
// Stage 3b: WMMA MLP. 8 waves/block; each wave owns 16 pixels end-to-end.
// A layout (16x32 f16): lanes 0-15: M=lane, halves 0-15 -> K kb+0..15 ;
//   lanes 16-31: M=lane-16, halves 0-15 -> kb+16..31.
// ---------------------------------------------------------------------------
__device__ inline v16h loadA(const _Float16* p) {
  HU u; u.h[0] = *(const v8h*)p; u.h[1] = *(const v8h*)(p + 8); return u.v;
}
__device__ inline v16h loadB(const _Float16* p) {
  HU u; u.h[0] = *(const v8h*)p; u.h[1] = *(const v8h*)(p + 8);  return u.v;
}

template<int KSIN, int KSHID, int NT>
__device__ inline const _Float16* do_layer(
    const _Float16* __restrict__ Ag,   // global mlp_input rows for this wave
    const _Float16* H,                 // wave's 16x128 hidden strip in LDS
    const _Float16* __restrict__ wp,   // B-fragment stream cursor
    const float* __restrict__ bias,
    _Float16* Hout, int lane)
{
  constexpr int KS = KSIN + KSHID;
  const int sel = lane >> 4, n = lane & 15;
  asm volatile("s_wait_dscnt 0" ::: "memory");   // prev layer's LDS stores visible
  v16h A[KS];
  #pragma unroll
  for (int s = 0; s < KSIN; ++s)
    A[s] = loadA(Ag + n*KPAD_IN + s*32 + sel*16);
  #pragma unroll
  for (int s = 0; s < KSHID; ++s)
    A[KSIN + s] = loadA(H + n*WIDTH + s*32 + sel*16);
  #pragma unroll
  for (int t = 0; t < NT; ++t) {
    v8f acc = {0.f,0.f,0.f,0.f,0.f,0.f,0.f,0.f};
    #pragma unroll
    for (int s = 0; s < KS; ++s) {
      v16h B = loadB(wp + ((size_t)(t*KS + s)*32 + lane)*16);
      acc = __builtin_amdgcn_wmma_f32_16x16x32_f16(
              false, A[s], false, B, (short)0, acc, false, false);
    }
    float bv = bias[t*16 + n];
    #pragma unroll
    for (int r = 0; r < 8; ++r) {
      float x = fmaxf(acc[r] + bv, 0.0f);
      Hout[(sel*8 + r)*WIDTH + t*16 + n] = (_Float16)x;   // D: M=r(+8), N=lane&15
    }
  }
  return wp + (size_t)NT * KS * 512;
}

__global__ __launch_bounds__(256) void k_mlp(
    const _Float16* __restrict__ mlpin, const _Float16* __restrict__ frags,
    const float* __restrict__ biases, float* __restrict__ out)
{
  __shared__ alignas(16) _Float16 hid[8][16][WIDTH];   // 32 KB
  const int lane    = threadIdx.x & 31;
  const int wave    = threadIdx.x >> 5;
  const int rowbase = blockIdx.x * 128 + wave * 16;
  const _Float16* Ag = mlpin + (size_t)rowbase * KPAD_IN;
  _Float16* H = &hid[wave][0][0];
  const _Float16* wp = frags;

  wp = do_layer<3,0,8>(Ag, H, wp, biases +   0, H, lane);  // L0: K=96
  wp = do_layer<3,4,8>(Ag, H, wp, biases + 128, H, lane);  // L1 skip: K=96+128
  wp = do_layer<3,4,8>(Ag, H, wp, biases + 256, H, lane);  // L2 skip
  wp = do_layer<0,4,8>(Ag, H, wp, biases + 384, H, lane);  // L3
  wp = do_layer<3,4,8>(Ag, H, wp, biases + 512, H, lane);  // L4 skip
  wp = do_layer<0,4,8>(Ag, H, wp, biases + 640, H, lane);  // L5
  wp = do_layer<0,4,8>(Ag, H, wp, biases + 768, H, lane);  // L6
  wp = do_layer<0,4,8>(Ag, H, wp, biases + 896, H, lane);  // L7

  // RGB head: 128 -> 3 (one 16-col N-tile, cols 0..2 valid), then sigmoid.
  const int sel = lane >> 4, n = lane & 15;
  asm volatile("s_wait_dscnt 0" ::: "memory");
  v16h A[4];
  #pragma unroll
  for (int s = 0; s < 4; ++s) A[s] = loadA(H + n*WIDTH + s*32 + sel*16);
  v8f acc = {0.f,0.f,0.f,0.f,0.f,0.f,0.f,0.f};
  #pragma unroll
  for (int s = 0; s < 4; ++s) {
    v16h B = loadB(wp + ((size_t)s*32 + lane)*16);
    acc = __builtin_amdgcn_wmma_f32_16x16x32_f16(
            false, A[s], false, B, (short)0, acc, false, false);
  }
  if (n < 3) {
    float bv = biases[1024 + n];
    #pragma unroll
    for (int r = 0; r < 8; ++r) {
      float x = acc[r] + bv;
      out[(size_t)(rowbase + sel*8 + r)*3 + n] = 1.0f / (1.0f + expf(-x));
    }
  }
}

// ---------------------------------------------------------------------------
extern "C" void kernel_launch(void* const* d_in, const int* in_sizes, int n_in,
                              void* d_out, int out_size, void* d_ws, size_t ws_size,
                              hipStream_t stream)
{
  const float* verts = (const float*)d_in[0];
  const int*   pvi   = (const int*)  d_in[1];
  const float* bary  = (const float*)d_in[2];
  const float* grid  = (const float*)d_in[3];
  const float* w[8]; const float* b[8];
  for (int i = 0; i < 8; ++i) {
    w[i] = (const float*)d_in[4 + i];
    b[i] = (const float*)d_in[12 + i];
  }
  const float* rgbw = (const float*)d_in[20];
  const float* rgbb = (const float*)d_in[21];
  float* out = (float*)d_out;

  char* ws = (char*)d_ws;
  float*    vfeat   = (float*)ws;                                  // 12,800,000 B
  _Float16* mlpin   = (_Float16*)(ws + 12800000);                  // 50,331,648 B
  _Float16* frags   = (_Float16*)(ws + 12800000 + 50331648);       //    331,776 B
  float*    biasbuf = (float*)   (ws + 12800000 + 50331648 + 331776); // 4,108 B

  k_vertex_feat<<<(N_VERTS + 255)/256, 256, 0, stream>>>(verts, grid, vfeat);
  k_pixel_input<<<N_PIX/256, 256, 0, stream>>>(verts, pvi, bary, vfeat, mlpin);
  k_prepack<<<dim3(7, 9), 256, 0, stream>>>(w[0],w[1],w[2],w[3],w[4],w[5],w[6],w[7],
                                            rgbw, frags);
  k_biases<<<5, 256, 0, stream>>>(b[0],b[1],b[2],b[3],b[4],b[5],b[6],b[7],
                                  rgbb, biasbuf);
  k_mlp<<<N_PIX/128, 256, 0, stream>>>(mlpin, frags, biasbuf, out);
}
